// TriangleMultiplicationOutgoing_43130061586669
// MI455X (gfx1250) — compile-verified, matrix operand order checked by reading, not run on hardware
//
#include <hip/hip_runtime.h>
#include <hip/hip_bf16.h>

typedef __attribute__((ext_vector_type(16))) _Float16 v16h;
typedef __attribute__((ext_vector_type(8)))  _Float16 v8h;
typedef __attribute__((ext_vector_type(4)))  _Float16 v4h;
typedef __attribute__((ext_vector_type(8)))  float    v8f;

union Frag16 { v16h v; v8h h[2]; };

constexpr int    NSEQ  = 512;
constexpr int    CDIM  = 128;
constexpr size_t PLANE = (size_t)NSEQ * NSEQ;   // 262144 positions

#define WMMA_F16(a, b, c) \
    __builtin_amdgcn_wmma_f32_16x16x32_f16(false, (a), false, (b), (short)0, (c), false, false)

__device__ __forceinline__ float sigmoidf_(float x) {
    return 1.0f / (1.0f + __expf(-x));
}

// Low 32 bits of a flat shared pointer == LDS byte address (ISA §10.2 aperture).
__device__ __forceinline__ uint32_t lds_off_(const void* p) {
    return (uint32_t)(uintptr_t)p;
}

// One per-lane 16B async copy: global (64-bit VGPR pair) -> LDS (32-bit VGPR).
__device__ __forceinline__ void async_b128_(uint32_t lds_addr, const void* gaddr) {
    asm volatile("global_load_async_to_lds_b128 %0, %1, off"
                 :: "v"(lds_addr), "v"(gaddr) : "memory");
}

// ---------------------------------------------------------------------------
// Kernel 0: convert six 128x128 f32 weight matrices into f16 WMMA B-fragment
// layout: frag[((mat*4+kc)*8+nt)*32 + lane] is one v16h (32B) per lane.
// Element e of lane l: K = kc*32 + 8*(l/16) + (e<8 ? e : e+8), N = nt*16 + l%16
// ---------------------------------------------------------------------------
__global__ __launch_bounds__(256) void k_prep_wfrag(
    const float* __restrict__ Wi,  const float* __restrict__ Wis,
    const float* __restrict__ Wj,  const float* __restrict__ Wjs,
    const float* __restrict__ Ws,  const float* __restrict__ Wp,
    _Float16* __restrict__ frag)
{
    int t = blockIdx.x * blockDim.x + threadIdx.x;
    if (t >= 6 * 4 * 8 * 32) return;
    int lane = t & 31;
    int nt   = (t >> 5) & 7;
    int kc   = (t >> 8) & 3;
    int mat  = t >> 10;
    const float* W = (mat == 0) ? Wi : (mat == 1) ? Wis : (mat == 2) ? Wj
                    : (mat == 3) ? Wjs : (mat == 4) ? Ws : Wp;
    int n     = nt * 16 + (lane & 15);
    int kbase = kc * 32 + 8 * (lane >> 4);
    v16h v;
    for (int e = 0; e < 16; ++e) {
        int k = kbase + (e < 8 ? e : e + 8);
        v[e] = (_Float16)W[k * CDIM + n];
    }
    ((v16h*)frag)[t] = v;
}

// ---------------------------------------------------------------------------
// Kernel 1: LayerNorm(x) -> i = (xWi+bi)*sig(xWis+bis)*mask (f16, planar [c][row])
//                          j likewise; sgate = sig(xWs+bs) (f16, [pos][c])
// Block: 256 threads = 8 waves, 128 rows per block.
// ---------------------------------------------------------------------------
__global__ __launch_bounds__(256) void k_ln_proj(
    const float* __restrict__ x2d, const float* __restrict__ mask,
    const float* __restrict__ g1,  const float* __restrict__ b1,
    const float* __restrict__ bi,  const float* __restrict__ bis,
    const float* __restrict__ bj,  const float* __restrict__ bjs,
    const float* __restrict__ bs,
    const _Float16* __restrict__ wfrag,
    _Float16* __restrict__ iT, _Float16* __restrict__ jT,
    _Float16* __restrict__ sgate)
{
    constexpr int LNS = 136;                       // 272B row stride: 4-bank shift/row
    __shared__ __align__(16) _Float16 lnx[128 * LNS];
    __shared__ float lmask[128];

    const int tid = threadIdx.x;
    const int r0  = blockIdx.x * 128;

    // ---- Phase 1: per-row LayerNorm (threads 0..127 own one row each) ----
    if (tid < 128) {
        const int row = r0 + tid;
        const float* xr = x2d + (size_t)row * CDIM;
        float s = 0.f, s2 = 0.f;
        for (int k = 0; k < CDIM; k += 4) {
            float4 v = *(const float4*)(xr + k);
            s  += v.x + v.y + v.z + v.w;
            s2 += v.x * v.x + v.y * v.y + v.z * v.z + v.w * v.w;
        }
        const float mu  = s * (1.0f / CDIM);
        const float var = s2 * (1.0f / CDIM) - mu * mu;
        const float rs  = rsqrtf(var + 1e-5f);
        for (int k = 0; k < CDIM; k += 4) {
            float4 v  = *(const float4*)(xr + k);
            float4 gv = *(const float4*)(g1 + k);
            float4 bv = *(const float4*)(b1 + k);
            v4h h;
            h[0] = (_Float16)((v.x - mu) * rs * gv.x + bv.x);
            h[1] = (_Float16)((v.y - mu) * rs * gv.y + bv.y);
            h[2] = (_Float16)((v.z - mu) * rs * gv.z + bv.z);
            h[3] = (_Float16)((v.w - mu) * rs * gv.w + bv.w);
            *(v4h*)&lnx[tid * LNS + k] = h;
        }
        lmask[tid] = mask[row];
    }
    __syncthreads();

    // ---- Phase 2: WMMA projections; wave w owns rows [w*16, w*16+16) ----
    const int w    = tid >> 5, l = tid & 31;
    const int half = l >> 4,  lc = l & 15;
    const int mrow = w * 16;

    Frag16 afr[4];
    for (int kc = 0; kc < 4; ++kc) {
        const _Float16* p = &lnx[(mrow + lc) * LNS + kc * 32 + 8 * half];
        afr[kc].h[0] = *(const v8h*)p;
        afr[kc].h[1] = *(const v8h*)(p + 16);
    }
    float mk[8];
    for (int r = 0; r < 8; ++r) mk[r] = lmask[mrow + 8 * half + r];

    const v16h* wf = (const v16h*)wfrag;

    auto gated_pair = [&](int matA, int matB, const float* __restrict__ bA,
                          const float* __restrict__ bB, _Float16* __restrict__ dst) {
        for (int nt = 0; nt < 8; ++nt) {
            v8f a0 = {}, a1 = {};
            for (int kc = 0; kc < 4; ++kc) {
                v16h bw0 = wf[((matA * 4 + kc) * 8 + nt) * 32 + l];
                v16h bw1 = wf[((matB * 4 + kc) * 8 + nt) * 32 + l];
                a0 = WMMA_F16(afr[kc].v, bw0, a0);
                a1 = WMMA_F16(afr[kc].v, bw1, a1);
            }
            const int c  = nt * 16 + lc;
            const float ba = bA[c], bb = bB[c];
            v8h pk;
            for (int r = 0; r < 8; ++r) {
                float val = (a0[r] + ba) * sigmoidf_(a1[r] + bb) * mk[r];
                pk[r] = (_Float16)val;
            }
            *(v8h*)(dst + (size_t)c * PLANE + r0 + mrow + 8 * half) = pk;
        }
    };
    gated_pair(0, 1, bi,  bis, iT);
    gated_pair(2, 3, bj,  bjs, jT);

    // Ws sigmoid gate, stored [pos][c] f16
    for (int nt = 0; nt < 8; ++nt) {
        v8f a0 = {};
        for (int kc = 0; kc < 4; ++kc)
            a0 = WMMA_F16(afr[kc].v, wf[((4 * 4 + kc) * 8 + nt) * 32 + l], a0);
        const int c = nt * 16 + lc;
        const float bsv = bs[c];
        for (int r = 0; r < 8; ++r) {
            size_t pos = (size_t)(r0 + mrow + 8 * half + r);
            sgate[pos * CDIM + c] = (_Float16)sigmoidf_(a0[r] + bsv);
        }
    }
}

// ---------------------------------------------------------------------------
// Kernel 2: per-channel triangle GEMM: tri[c][i][j] = sum_k iT[c][i][k]*jT[c][j][k]
// Block = 128x128 output tile of one channel; 8 waves, each 16(M)x128(N).
// Panels staged via double-buffered GLOBAL_LOAD_ASYNC_TO_LDS_B128 (ASYNCcnt),
// LDS rows padded to 40 halfs (80B) -> conflict-free b128 fragment reads.
// ---------------------------------------------------------------------------
__global__ __launch_bounds__(256) void k_tri(
    const _Float16* __restrict__ iT, const _Float16* __restrict__ jT,
    float* __restrict__ tri)
{
    constexpr int KSTR = 40;                       // 80B row stride
    __shared__ __align__(16) _Float16 As[2][128 * KSTR];
    __shared__ __align__(16) _Float16 Bs[2][128 * KSTR];

    const int tid = threadIdx.x;
    const int c   = blockIdx.x >> 4;
    const int t4  = blockIdx.x & 15;
    const int i0  = (t4 >> 2) * 128;
    const int j0  = (t4 & 3) * 128;

    const _Float16* Ag = iT + (size_t)c * PLANE + (size_t)i0 * NSEQ;
    const _Float16* Bg = jT + (size_t)c * PLANE + (size_t)j0 * NSEQ;

    const int w = tid >> 5, l = tid & 31, half = l >> 4, lc = l & 15;

    // loader: each lane owns two 16B chunks per panel (elements tid, tid+256)
    const int r0e = tid >> 2, q0 = tid & 3;
    const int r1e = (tid + 256) >> 2;              // q1 == q0
    const size_t gA0 = (size_t)r0e * NSEQ + q0 * 8;
    const size_t gA1 = (size_t)r1e * NSEQ + q0 * 8;

    uint32_t lA0[2], lA1[2], lB0[2], lB1[2];
    for (int b = 0; b < 2; ++b) {
        lA0[b] = lds_off_(&As[b][r0e * KSTR + q0 * 8]);
        lA1[b] = lds_off_(&As[b][r1e * KSTR + q0 * 8]);
        lB0[b] = lds_off_(&Bs[b][r0e * KSTR + q0 * 8]);
        lB1[b] = lds_off_(&Bs[b][r1e * KSTR + q0 * 8]);
    }

    auto issue_panel = [&](int b, int ks) {
        async_b128_(lA0[b], Ag + gA0 + ks);
        async_b128_(lA1[b], Ag + gA1 + ks);
        async_b128_(lB0[b], Bg + gA0 + ks);
        async_b128_(lB1[b], Bg + gA1 + ks);
    };

    issue_panel(0, 0);                              // prologue: panel 0 in flight

    v8f acc[8] = {};

    for (int kb = 0; kb < 16; ++kb) {
        if (kb < 15) {
            issue_panel((kb + 1) & 1, (kb + 1) * 32);   // queue next panel
            // current panel's 4 ops retired; next panel's 4 may remain in flight
            asm volatile("s_wait_asynccnt 0x4" ::: "memory");
        } else {
            asm volatile("s_wait_asynccnt 0x0" ::: "memory");
        }
        __syncthreads();                            // all waves' panel landed

        const _Float16* Ab = As[kb & 1];
        const _Float16* Bb = Bs[kb & 1];
        Frag16 a;
        const _Float16* ap = &Ab[(w * 16 + lc) * KSTR + 8 * half];
        a.h[0] = *(const v8h*)ap;
        a.h[1] = *(const v8h*)(ap + 16);
        for (int nt = 0; nt < 8; ++nt) {
            Frag16 b;
            const _Float16* bp_ = &Bb[(nt * 16 + lc) * KSTR + 8 * half];
            b.h[0] = *(const v8h*)bp_;
            b.h[1] = *(const v8h*)(bp_ + 16);
            acc[nt] = WMMA_F16(a.v, b.v, acc[nt]);
        }
        __syncthreads();                            // buffer safe to overwrite
    }

    float* out = tri + (size_t)c * PLANE;
    const int irow = i0 + w * 16 + 8 * half;
    for (int nt = 0; nt < 8; ++nt) {
        const int j = j0 + nt * 16 + lc;
        for (int r = 0; r < 8; ++r)
            out[(size_t)(irow + r) * NSEQ + j] = acc[nt][r];
    }
}

// ---------------------------------------------------------------------------
// Kernel 3: LayerNorm over c (from planar tri), @Wp + bp, * sgate, * mask
// Block handles 64 positions; output [pos][c] f32 (required layout).
// ---------------------------------------------------------------------------
__global__ __launch_bounds__(256) void k_out(
    const float* __restrict__ mask, const float* __restrict__ g2,
    const float* __restrict__ b2,   const float* __restrict__ bp,
    const _Float16* __restrict__ sgate, const _Float16* __restrict__ wfrag,
    const float* __restrict__ tri, float* __restrict__ out)
{
    constexpr int RSTR = 68;                        // f32 [c][p] stride (272B)
    constexpr int LNS  = 136;                       // f16 [p][c] stride (272B)
    __shared__ __align__(16) float    raw[128 * RSTR];
    __shared__ __align__(16) _Float16 lnx[64 * LNS];
    __shared__ float lmask[64];

    const int    tid = threadIdx.x;
    const size_t p0  = (size_t)blockIdx.x * 64;

    // Phase 1: load planar tri tile [128c x 64p] (coalesced per channel row)
    {
        const int cch = tid >> 1, seg = tid & 1;
        const float* src = tri + (size_t)cch * PLANE + p0 + seg * 32;
        float* dst = &raw[cch * RSTR + seg * 32];
        for (int q = 0; q < 32; q += 4)
            *(float4*)(dst + q) = *(const float4*)(src + q);
    }
    if (tid < 64) lmask[tid] = mask[p0 + tid];
    __syncthreads();

    // Phase 2: per-position stats + normalize -> f16 [p][c]
    if (tid < 64) {
        float s = 0.f, s2 = 0.f;
        for (int cc = 0; cc < CDIM; ++cc) {
            float v = raw[cc * RSTR + tid];
            s += v; s2 += v * v;
        }
        const float mu  = s * (1.0f / CDIM);
        const float var = s2 * (1.0f / CDIM) - mu * mu;
        const float rs  = rsqrtf(var + 1e-5f);
        for (int cc = 0; cc < CDIM; ++cc) {
            float v = (raw[cc * RSTR + tid] - mu) * rs * g2[cc] + b2[cc];
            lnx[tid * LNS + cc] = (_Float16)v;
        }
    }
    __syncthreads();

    // Phase 3: WMMA x WpFrag (mat index 5). 8 waves: (w&3)->M tile, (w>>2)->N half
    const int w = tid >> 5, l = tid & 31, half = l >> 4, lc = l & 15;
    const int mbase = (w & 3) * 16;
    const int ntb   = (w >> 2) * 4;

    Frag16 afr[4];
    for (int kc = 0; kc < 4; ++kc) {
        const _Float16* p = &lnx[(mbase + lc) * LNS + kc * 32 + 8 * half];
        afr[kc].h[0] = *(const v8h*)p;
        afr[kc].h[1] = *(const v8h*)(p + 16);
    }
    const v16h* wf = (const v16h*)wfrag;

    for (int nti = 0; nti < 4; ++nti) {
        const int nt = ntb + nti;
        v8f a0 = {};
        for (int kc = 0; kc < 4; ++kc)
            a0 = WMMA_F16(afr[kc].v, wf[((5 * 4 + kc) * 8 + nt) * 32 + l], a0);
        const int cc  = nt * 16 + lc;
        const float bpv = bp[cc];
        for (int r = 0; r < 8; ++r) {
            const int    pl = mbase + 8 * half + r;
            const size_t p  = p0 + pl;
            float gate = (float)sgate[p * CDIM + cc];
            out[p * CDIM + cc] = (a0[r] + bpv) * gate * lmask[pl];
        }
    }
}

// ---------------------------------------------------------------------------
extern "C" void kernel_launch(void* const* d_in, const int* in_sizes, int n_in,
                              void* d_out, int out_size, void* d_ws, size_t ws_size,
                              hipStream_t stream)
{
    const float* x2d  = (const float*)d_in[0];
    const float* mask = (const float*)d_in[1];
    const float* g1   = (const float*)d_in[2];
    const float* b1   = (const float*)d_in[3];
    const float* g2   = (const float*)d_in[4];
    const float* b2   = (const float*)d_in[5];
    const float* Wi   = (const float*)d_in[6];
    const float* bi   = (const float*)d_in[7];
    const float* Wis  = (const float*)d_in[8];
    const float* bis  = (const float*)d_in[9];
    const float* Wj   = (const float*)d_in[10];
    const float* bj   = (const float*)d_in[11];
    const float* Wjs  = (const float*)d_in[12];
    const float* bjs  = (const float*)d_in[13];
    const float* Wp   = (const float*)d_in[14];
    const float* bp   = (const float*)d_in[15];
    const float* Ws   = (const float*)d_in[16];
    const float* bs   = (const float*)d_in[17];

    char* ws = (char*)d_ws;
    const size_t half_plane_all = PLANE * CDIM * sizeof(_Float16);  // 64 MiB
    _Float16* iT    = (_Float16*)(ws);
    _Float16* jT    = (_Float16*)(ws + half_plane_all);
    _Float16* sgate = (_Float16*)(ws + 2 * half_plane_all);
    float*    tri   = (float*)   (ws + 3 * half_plane_all);
    _Float16* wfrag = (_Float16*)(ws + 3 * half_plane_all + PLANE * CDIM * sizeof(float));

    k_prep_wfrag<<<24, 256, 0, stream>>>(Wi, Wis, Wj, Wjs, Ws, Wp, wfrag);
    k_ln_proj<<<(int)(PLANE / 128), 256, 0, stream>>>(
        x2d, mask, g1, b1, bi, bis, bj, bjs, bs, wfrag, iT, jT, sgate);
    k_tri<<<CDIM * 16, 256, 0, stream>>>(iT, jT, tri);
    k_out<<<(int)(PLANE / 64), 256, 0, stream>>>(
        mask, g2, b2, bp, sgate, wfrag, tri, (float*)d_out);
}
